// GNN_37761352466454
// MI455X (gfx1250) — matrix operand order repaired and checked by reading, not compiled, not run on hardware
//
#include <hip/hip_runtime.h>
#include <math.h>

typedef __attribute__((ext_vector_type(16))) _Float16 v16h;
typedef __attribute__((ext_vector_type(8)))  float    v8f;

#define FD 128   // feature dim
#define MT 4     // M sub-tiles per block (64 rows)

// ---------------------------------------------------------------------------
// Dense GEMM: Out[nrows x 128] = act(A) @ W, act(x) = relu(x + bias[k]) when
// bias != nullptr (fused epilogue of the PREVIOUS layer's aggregation).
//
// Block = 256 threads (8 waves) owns a 64-row x 128-col output tile.
//  - A tile (64x128) staged in LDS as f16 (bias+ReLU fused on the fly)
//  - W staged TRANSPOSED in LDS as f16 so B-fragment gathers are contiguous
//    (k runs fastest) and merge into ds_load_b128 pairs
//  - each wave owns a 16-col slice; per k-step it loads one B fragment and
//    reuses it across 4 independent WMMA accumulators (no D->A/B hazards)
// 16 v_wmma_f32_16x16x32_f16 per block pass; K = 128 = 4 steps.
// ---------------------------------------------------------------------------
__global__ __launch_bounds__(256)
void gcn_gemm_wmma(const float* __restrict__ A, const float* __restrict__ W,
                   const float* __restrict__ bias, float* __restrict__ Out,
                   int nrows)
{
    __shared__ _Float16 As[MT * 16 * FD];   // 16 KB: 64 x 128 activations
    __shared__ _Float16 Ws[FD * FD];        // 32 KB: W transposed, [n][k]

    const int tid   = threadIdx.x;
    const int lane  = tid & 31;
    const int wv    = tid >> 5;             // wave id 0..7 -> 16-col tile
    const int mbase = blockIdx.x * (MT * 16);

    // Stage A tile (64x128) as f16 with fused bias+ReLU.
    for (int i = tid; i < MT * 16 * FD; i += 256) {
        const int r   = i >> 7;
        const int k   = i & (FD - 1);
        const int row = mbase + r;
        float v = (row < nrows) ? A[(size_t)row * FD + k] : 0.0f;
        if (bias) v = fmaxf(v + bias[k], 0.0f);
        As[i] = (_Float16)v;
    }
    // Stage W transposed: Ws[n*128 + k] = W[k*128 + n]  (f16)
    for (int i = tid; i < FD * FD; i += 256) {
        const int n = i >> 7;
        const int k = i & (FD - 1);
        Ws[i] = (_Float16)W[(size_t)k * FD + n];
    }
    __syncthreads();

    const int nb = wv * 16;                 // output column base
    const int hi = lane >> 4;               // lane group (0/1)
    const int ln = lane & 15;

    v8f c[MT] = {};
    #pragma unroll
    for (int kk = 0; kk < FD; kk += 32) {
        // B fragment: col = nb+ln, K = kk + e (+8 for e>=8) (+8 for hi lanes) — two contiguous runs
        v16h b = {};
        #pragma unroll
        for (int e = 0; e < 16; ++e)
            b[e] = Ws[(nb + ln) * FD + kk + e + ((e >> 3) << 3) + (hi << 3)];

        #pragma unroll
        for (int m = 0; m < MT; ++m) {
            // A fragment: row = m*16 + (lane&15),
            // K = kk + e (+8 for e>=8) (+8 for hi lanes) — two contiguous runs
            v16h a = {};
            #pragma unroll
            for (int e = 0; e < 16; ++e)
                a[e] = As[(m * 16 + ln) * FD + kk + e + ((e >> 3) << 3) + (hi << 3)];
            c[m] = __builtin_amdgcn_wmma_f32_16x16x32_f16(
                       false, a, false, b, (short)0, c[m], false, false);
        }
    }

    // C/D layout: VGPR r -> M = r (+8 for hi lanes), N = lane&15
    #pragma unroll
    for (int m = 0; m < MT; ++m) {
        #pragma unroll
        for (int r = 0; r < 8; ++r) {
            const int row = mbase + m * 16 + r + (hi << 3);
            if (row < nrows) Out[(size_t)row * FD + nb + ln] = c[m][r];
        }
    }
}

// ---------------------------------------------------------------------------
// Edge scatter: agg[dst] += w * pre[src], 128 floats per edge.
// One wave per edge; each lane handles a float4 slice; L2 f32 atomics.
// ---------------------------------------------------------------------------
__global__ __launch_bounds__(256)
void edge_scatter_vec(const int* __restrict__ src, const int* __restrict__ dst,
                      const float* __restrict__ w, const float* __restrict__ pre,
                      float* __restrict__ agg, int E)
{
    const int t = blockIdx.x * blockDim.x + threadIdx.x;
    const int e = t >> 5;
    const int q = t & 31;
    if (e >= E) return;
    const int   s  = src[e];
    const int   d  = dst[e];
    const float wt = w[e];
    const float4 v = ((const float4*)(pre + (size_t)s * FD))[q];
    float* ap = agg + (size_t)d * FD + q * 4;
    atomicAdd(ap + 0, v.x * wt);
    atomicAdd(ap + 1, v.y * wt);
    atomicAdd(ap + 2, v.z * wt);
    atomicAdd(ap + 3, v.w * wt);
}

// ---------------------------------------------------------------------------
// Final projection: hf[i] = relu(agg[i] + b2) . Wf   (one wave per node)
// ---------------------------------------------------------------------------
__global__ __launch_bounds__(256)
void node_dot(const float* __restrict__ agg, const float* __restrict__ bias,
              const float* __restrict__ Wf, float* __restrict__ hf, int nrows)
{
    const int t    = blockIdx.x * blockDim.x + threadIdx.x;
    const int node = t >> 5;
    const int lane = t & 31;
    if (node >= nrows) return;
    const float* row = agg + (size_t)node * FD;
    float s = 0.0f;
    #pragma unroll
    for (int i = 0; i < 4; ++i) {
        const int f = lane * 4 + i;
        s += fmaxf(row[f] + bias[f], 0.0f) * Wf[f];
    }
    #pragma unroll
    for (int off = 16; off > 0; off >>= 1)
        s += __shfl_down(s, off, 32);
    if (lane == 0) hf[node] = s;
}

// Scalar edge scatter for the 1-wide final layer.
__global__ __launch_bounds__(256)
void edge_scatter_scalar(const int* __restrict__ src, const int* __restrict__ dst,
                         const float* __restrict__ w, const float* __restrict__ hf,
                         float* __restrict__ aggf, int E)
{
    const int e = blockIdx.x * blockDim.x + threadIdx.x;
    if (e < E) atomicAdd(&aggf[dst[e]], w[e] * hf[src[e]]);
}

// Per-graph mean of sigmoid(aggf + bf), driven by segment_ids.
__global__ __launch_bounds__(256)
void readout(const float* __restrict__ aggf, const float* __restrict__ bf,
             const int* __restrict__ seg, float* __restrict__ sums,
             float* __restrict__ cnts, int nrows)
{
    const int i = blockIdx.x * blockDim.x + threadIdx.x;
    if (i >= nrows) return;
    const float x = aggf[i] + bf[0];
    const float sgm = 1.0f / (1.0f + expf(-x));
    atomicAdd(&sums[seg[i]], sgm);
    atomicAdd(&cnts[seg[i]], 1.0f);
}

__global__ __launch_bounds__(128)
void finalize(const float* __restrict__ sums, const float* __restrict__ cnts,
              float* __restrict__ out, int G)
{
    const int g = blockIdx.x * blockDim.x + threadIdx.x;
    if (g < G) out[g] = sums[g] / cnts[g];
}

// ---------------------------------------------------------------------------
extern "C" void kernel_launch(void* const* d_in, const int* in_sizes, int n_in,
                              void* d_out, int out_size, void* d_ws, size_t ws_size,
                              hipStream_t stream)
{
    const float* X  = (const float*)d_in[0];
    const float* ew = (const float*)d_in[1];
    const float* W1 = (const float*)d_in[2];
    const float* b1 = (const float*)d_in[3];
    const float* W2 = (const float*)d_in[4];
    const float* b2 = (const float*)d_in[5];
    const float* Wf = (const float*)d_in[6];
    const float* bf = (const float*)d_in[7];
    const int*  eix = (const int*)d_in[8];
    const int*  seg = (const int*)d_in[9];

    const int Nn = in_sizes[0] / FD;   // 50000 nodes
    const int E  = in_sizes[1];        // 800000 edges
    const int G  = out_size;           // 100 graphs
    const int* src = eix;
    const int* dst = eix + E;

    // Workspace layout (~52 MB; fits comfortably in the 192 MB L2)
    float* pre  = (float*)d_ws;                 // [N, 128] GEMM output
    float* agg  = pre  + (size_t)Nn * FD;       // [N, 128] scatter accumulator
    float* hf   = agg  + (size_t)Nn * FD;       // [N]
    float* aggf = hf   + Nn;                    // [N]
    float* sums = aggf + Nn;                    // [G]
    float* cnts = sums + G;                     // [G]

    const int mtiles  = (Nn + MT * 16 - 1) / (MT * 16);  // 782 row tiles
    const int egrid32 = (E * 32 + 255) / 256;            // wave-per-edge
    const int egrid1  = (E + 255) / 256;
    const int ngrid32 = (Nn * 32 + 255) / 256;
    const int ngrid1  = (Nn + 255) / 256;

    // Layer 1: pre = X @ W1 ; agg1 = scatter(pre)
    gcn_gemm_wmma<<<mtiles, 256, 0, stream>>>(X, W1, nullptr, pre, Nn);
    hipMemsetAsync(agg, 0, (size_t)Nn * FD * sizeof(float), stream);
    edge_scatter_vec<<<egrid32, 256, 0, stream>>>(src, dst, ew, pre, agg, E);

    // Layer 2: pre = relu(agg1 + b1) @ W2  (activation fused in A staging)
    gcn_gemm_wmma<<<mtiles, 256, 0, stream>>>(agg, W2, b1, pre, Nn);
    hipMemsetAsync(agg, 0, (size_t)Nn * FD * sizeof(float), stream);
    edge_scatter_vec<<<egrid32, 256, 0, stream>>>(src, dst, ew, pre, agg, E);

    // Final layer: hf = relu(agg2 + b2) . Wf ; aggf = scatter(hf)
    node_dot<<<ngrid32, 256, 0, stream>>>(agg, b2, Wf, hf, Nn);
    hipMemsetAsync(aggf, 0, (size_t)Nn * sizeof(float), stream);
    edge_scatter_scalar<<<egrid1, 256, 0, stream>>>(src, dst, ew, hf, aggf, E);

    // Readout: per-graph mean of sigmoid(aggf + bf)
    hipMemsetAsync(sums, 0, (size_t)2 * G * sizeof(float), stream);
    readout<<<ngrid1, 256, 0, stream>>>(aggf, bf, seg, sums, cnts, Nn);
    finalize<<<(G + 127) / 128, 128, 0, stream>>>(sums, cnts, (float*)d_out, G);
}
